// _BinaryLinear_52802327937389
// MI455X (gfx1250) — matrix operand order repaired and checked by reading, not compiled, not run on hardware
//
#include <hip/hip_runtime.h>

// ---------------------------------------------------------------------------
// BinaryLinear: out = x @ decode(W)^T,  W codes {0,1} -> {+1,-1}
//   x: [8192, 4096] fp32   W: [4096, 4096] int32   out: [8192, 4096] fp32
//
// Decode W once to bf16 +/-1 in d_ws (32 MB; whole matrix L2-resident).
// GEMM: 128x128 tiles, 8 waves, K-step 32. Each fp32 activation is split
// into hi+lo bf16 planes -> two v_wmma_f32_16x16x32_bf16 per tile into a
// shared f32 accumulator (~fp32 accuracy at bf16 matrix speed).
// B tiles are moved memory->LDS with GLOBAL_LOAD_ASYNC_TO_LDS_B128
// (ASYNCcnt), double-buffered LDS, one barrier per K-step. The fp32->bf16
// split is pure 32-bit VALU with v_perm_b32 packing (no 16-bit mov chains).
// ---------------------------------------------------------------------------

typedef __attribute__((ext_vector_type(16))) __bf16 v16bf;
typedef __attribute__((ext_vector_type(8)))  float  v8f;

#define TOKENS 8192
#define KDIM   4096
#define OUTCH  4096
#define BM 128
#define BN 128
#define BK 32
#define LDSK (BK + 8)   // row stride 80B = 20 banks -> conflict-free gathers

// ---- decode {0,1} int32 codes -> bf16 {+1,-1} bit patterns (8 elems/thread)
__global__ void __launch_bounds__(256)
decode_w(const int* __restrict__ w, unsigned short* __restrict__ wd) {
  long long gid = (long long)blockIdx.x * blockDim.x + threadIdx.x;
  const int4* wp = reinterpret_cast<const int4*>(w) + gid * 2;
  int4 c0 = wp[0];
  int4 c1 = wp[1];
  const unsigned int P = 0x3F80u, Nn = 0xBF80u;  // +1.0 / -1.0 bf16
  uint4 o;
  o.x = (c0.x == 1 ? P : Nn) | ((c0.y == 1 ? P : Nn) << 16);
  o.y = (c0.z == 1 ? P : Nn) | ((c0.w == 1 ? P : Nn) << 16);
  o.z = (c1.x == 1 ? P : Nn) | ((c1.y == 1 ? P : Nn) << 16);
  o.w = (c1.z == 1 ? P : Nn) | ((c1.w == 1 ? P : Nn) << 16);
  reinterpret_cast<uint4*>(wd)[gid] = o;
}

union FragBF { v16bf v; uint4 q[2]; };

// ---- 128x128 tile GEMM, 8 waves, split-bf16 WMMA, async B, double-buffer ----
__global__ void __launch_bounds__(256)
gemm_bin(const float* __restrict__ x, const unsigned short* __restrict__ wd,
         float* __restrict__ out) {
  __shared__ __attribute__((aligned(16))) unsigned short Ah[2][BM][LDSK];
  __shared__ __attribute__((aligned(16))) unsigned short Al[2][BM][LDSK];
  __shared__ __attribute__((aligned(16))) unsigned short Bs[2][BN][LDSK];

  const int tid  = threadIdx.x;
  const int lane = tid & 31;
  const int wv   = tid >> 5;        // wave 0..7
  const int wm   = wv & 3;          // 4 waves along M
  const int wn   = wv >> 2;         // 2 waves along N
  const int l15  = lane & 15;
  const int lh   = lane >> 4;       // lane half (CDNA5 fragment layout)

  const int m0 = blockIdx.y * BM;
  const int n0 = blockIdx.x * BN;

  // cooperative load mapping: 2 threads per row, 16 elems per thread
  const int row  = tid >> 1;        // 0..127
  const int half = tid & 1;        // which 16-elem chunk of the K-slab

  v8f acc[2][4];
  const v8f vzero = {0.f, 0.f, 0.f, 0.f, 0.f, 0.f, 0.f, 0.f};
#pragma unroll
  for (int mi = 0; mi < 2; ++mi)
#pragma unroll
    for (int ni = 0; ni < 4; ++ni) acc[mi][ni] = vzero;

  const float*          xg = x  + (size_t)(m0 + row) * KDIM + half * 16;
  const unsigned short* wg = wd + (size_t)(n0 + row) * KDIM + half * 16;

  float4 xr[4];

  // async: global B tile -> LDS, 32B per thread, IOFFSET applies to both sides
  auto issueB = [&](int buf, int kt) {
    unsigned lb = (unsigned)(size_t)&Bs[buf][row][half * 16];
    const unsigned short* g = wg + (size_t)kt * BK;
    asm volatile(
        "global_load_async_to_lds_b128 %0, %1, off\n\t"
        "global_load_async_to_lds_b128 %0, %1, off offset:16"
        :: "v"(lb), "v"(g) : "memory");
  };
  auto loadX = [&](int kt) {
    const float4* p = reinterpret_cast<const float4*>(xg + (size_t)kt * BK);
    xr[0] = p[0]; xr[1] = p[1]; xr[2] = p[2]; xr[3] = p[3];
  };
  // split fp32 regs into bf16 hi/lo planes; all 32-bit VALU, v_perm packing
  auto stageA = [&](int buf) {
    float f[16];
    f[0]  = xr[0].x; f[1]  = xr[0].y; f[2]  = xr[0].z; f[3]  = xr[0].w;
    f[4]  = xr[1].x; f[5]  = xr[1].y; f[6]  = xr[1].z; f[7]  = xr[1].w;
    f[8]  = xr[2].x; f[9]  = xr[2].y; f[10] = xr[2].z; f[11] = xr[2].w;
    f[12] = xr[3].x; f[13] = xr[3].y; f[14] = xr[3].z; f[15] = xr[3].w;
#pragma unroll
    for (int j = 0; j < 16; j += 2) {
      unsigned u0 = __float_as_uint(f[j]);
      unsigned u1 = __float_as_uint(f[j + 1]);
      // bf16 RNE: u + 0x7FFF + lsb(hi half)
      unsigned r0 = u0 + 0x7FFFu + ((u0 >> 16) & 1u);
      unsigned r1 = u1 + 0x7FFFu + ((u1 >> 16) & 1u);
      // pack the two high halfwords in one v_perm_b32
      unsigned hpack = __builtin_amdgcn_perm(r1, r0, 0x07060302u);
      // exact fp32 residual vs truncated-hi
      float l0 = f[j]     - __uint_as_float(r0 & 0xFFFF0000u);
      float l1 = f[j + 1] - __uint_as_float(r1 & 0xFFFF0000u);
      unsigned v0 = __float_as_uint(l0);
      unsigned v1 = __float_as_uint(l1);
      unsigned s0 = v0 + 0x7FFFu + ((v0 >> 16) & 1u);
      unsigned s1 = v1 + 0x7FFFu + ((v1 >> 16) & 1u);
      unsigned lpack = __builtin_amdgcn_perm(s1, s0, 0x07060302u);
      *reinterpret_cast<unsigned*>(&Ah[buf][row][half * 16 + j]) = hpack;
      *reinterpret_cast<unsigned*>(&Al[buf][row][half * 16 + j]) = lpack;
    }
  };

  // ---- prologue: stage tile 0 into buffer 0 ----
  issueB(0, 0);
  loadX(0);
  stageA(0);
  asm volatile("s_wait_asynccnt 0x0" ::: "memory");
  __syncthreads();

  const int NK = KDIM / BK;  // 128 K-steps
  for (int kt = 0; kt < NK; ++kt) {
    const int cur = kt & 1;
    const int nxt = cur ^ 1;
    const bool more = (kt + 1 < NK);

    // kick off next tile's traffic before the WMMA burst
    if (more) {
      issueB(nxt, kt + 1);
      loadX(kt + 1);
    }

    // gather fragments per CDNA5 16-bit WMMA layouts
    FragBF ah[2], al[2], fb[4];
#pragma unroll
    for (int mi = 0; mi < 2; ++mi) {
      const int m = wm * 32 + mi * 16 + l15;
      // A 16x32: lanes 0-15 hold K{0-7,16-23}, lanes 16-31 hold K{8-15,24-31}
      ah[mi].q[0] = *reinterpret_cast<const uint4*>(&Ah[cur][m][lh * 8]);
      ah[mi].q[1] = *reinterpret_cast<const uint4*>(&Ah[cur][m][16 + lh * 8]);
      al[mi].q[0] = *reinterpret_cast<const uint4*>(&Al[cur][m][lh * 8]);
      al[mi].q[1] = *reinterpret_cast<const uint4*>(&Al[cur][m][16 + lh * 8]);
    }
#pragma unroll
    for (int ni = 0; ni < 4; ++ni) {
      const int n = wn * 64 + ni * 16 + l15;
      // B 32x16: lane = column; lanes 0-15 hold K{0-7,16-23}, lanes 16-31 hold K{8-15,24-31}
      fb[ni].q[0] = *reinterpret_cast<const uint4*>(&Bs[cur][n][lh * 8]);
      fb[ni].q[1] = *reinterpret_cast<const uint4*>(&Bs[cur][n][16 + lh * 8]);
    }

#pragma unroll
    for (int mi = 0; mi < 2; ++mi)
#pragma unroll
      for (int ni = 0; ni < 4; ++ni) {
        acc[mi][ni] = __builtin_amdgcn_wmma_f32_16x16x32_bf16(
            false, ah[mi].v, false, fb[ni].v, (short)0, acc[mi][ni], false, false);
        acc[mi][ni] = __builtin_amdgcn_wmma_f32_16x16x32_bf16(
            false, al[mi].v, false, fb[ni].v, (short)0, acc[mi][ni], false, false);
      }

    // stage next x tile into the other buffer (overlaps WMMA issue window)
    if (more) stageA(nxt);

    asm volatile("s_wait_asynccnt 0x0" ::: "memory");
    __syncthreads();
  }

  // C/D layout: VGPR r -> row r (lanes 0-15) / row r+8 (lanes 16-31)
#pragma unroll
  for (int mi = 0; mi < 2; ++mi) {
#pragma unroll
    for (int ni = 0; ni < 4; ++ni) {
      const int rbase = m0 + wm * 32 + mi * 16 + lh * 8;
      const int col   = n0 + wn * 64 + ni * 16 + l15;
#pragma unroll
      for (int r = 0; r < 8; ++r)
        out[(size_t)(rbase + r) * OUTCH + col] = acc[mi][ni][r];
    }
  }
}

extern "C" void kernel_launch(void* const* d_in, const int* in_sizes, int n_in,
                              void* d_out, int out_size, void* d_ws, size_t ws_size,
                              hipStream_t stream) {
  const float* x = (const float*)d_in[0];
  const int*   w = (const int*)d_in[1];
  float*       out  = (float*)d_out;
  unsigned short* wdec = (unsigned short*)d_ws;   // 4096*4096*2 = 32 MB scratch

  // 1) decode {0,1} -> bf16 +/-1  (8 elems/thread)
  const int total = OUTCH * KDIM;
  decode_w<<<total / (256 * 8), 256, 0, stream>>>(w, wdec);

  // 2) split-bf16 WMMA GEMM, 128x128 tiles
  dim3 grid(OUTCH / BN, TOKENS / BM);   // (32, 64)
  gemm_bin<<<grid, 256, 0, stream>>>(x, wdec, out);
}
